// CausalSelfAttention_48404281426139
// MI455X (gfx1250) — matrix operand order repaired and checked by reading, not compiled, not run on hardware
//
#include <hip/hip_runtime.h>

// ---------------- problem constants ----------------
#define BB 2
#define TT 2048
#define CC 2048
#define NH 16
#define DH 128

typedef __attribute__((ext_vector_type(4)))  __bf16 v4bf;
typedef __attribute__((ext_vector_type(8)))  __bf16 v8bf;
typedef __attribute__((ext_vector_type(16))) __bf16 v16bf;
typedef __attribute__((ext_vector_type(8)))  float  v8f;

static_assert(sizeof(__bf16) == 2, "bf16 size");

// Build a 16-element bf16 A/B fragment from two 16-byte LDS/global chunks.
static __device__ __forceinline__ v16bf ld16x2(const __bf16* p0, const __bf16* p1) {
  v8bf a = *(const v8bf*)p0;
  v8bf b = *(const v8bf*)p1;
  return __builtin_shufflevector(a, b, 0,1,2,3,4,5,6,7,8,9,10,11,12,13,14,15);
}

static __device__ __forceinline__ v8f wmma_bf16(v16bf a, v16bf b, v8f c) {
  return __builtin_amdgcn_wmma_f32_16x16x32_bf16(false, a, false, b, (short)0, c, false, false);
}

// =====================================================================
// GEMM: C[M,N] = A[M,K] * B[K,N]   (bf16 WMMA, f32 accumulate)
// Block tile 128x128, K-tile 32, 256 threads = 8 waves in 2(M)x4(N).
// Double-buffered LDS pipeline: load(t+1) -> compute(t) -> store(t+1).
// =====================================================================
template<bool A_BF16, bool OUT_BF16>
__global__ __launch_bounds__(256) void gemm_wmma_k(
    const void* __restrict__ Aptr, const float* __restrict__ Bw,
    void* __restrict__ Cptr, int M, int N, int K, int lda, int ldb, int ldc) {
  __shared__ __align__(16) __bf16 As[2][128][40];   // [buf][m][k], padded rows
  __shared__ __align__(16) __bf16 Bs[2][128][40];   // [buf][n][k] transposed

  const int tid  = threadIdx.x;
  const int lane = tid & 31;
  const int wave = tid >> 5;
  const int wm   = wave >> 2;          // 0..1
  const int wn   = wave & 3;           // 0..3
  const int l16  = lane & 15;
  const int half = lane >> 4;          // 0/1
  const int m0   = blockIdx.y * 128;
  const int n0   = blockIdx.x * 128;

  int am[4], ak[4], bk[4], bn[4];
#pragma unroll
  for (int j = 0; j < 4; ++j) {
    int f = j * 1024 + tid * 4;
    am[j] = f >> 5;  ak[j] = f & 31;    // A: 128 x 32
    bk[j] = f >> 7;  bn[j] = f & 127;   // B: 32 x 128
  }

  v8f acc[4][2] = {};
  const int ntiles = K >> 5;

  float4 raf[4];  v4bf rab[4];  float4 rbf[4];

  for (int t = -1; t < ntiles; ++t) {
    const bool more = (t + 1) < ntiles;
    // ---- issue global loads for tile t+1 ----
    if (more) {
      const int k0 = (t + 1) << 5;
#pragma unroll
      for (int j = 0; j < 4; ++j) {
        if (A_BF16)
          rab[j] = *(const v4bf*)((const __bf16*)Aptr + (size_t)(m0 + am[j]) * lda + k0 + ak[j]);
        else
          raf[j] = *(const float4*)((const float*)Aptr + (size_t)(m0 + am[j]) * lda + k0 + ak[j]);
        rbf[j] = *(const float4*)(Bw + (size_t)(k0 + bk[j]) * ldb + n0 + bn[j]);
      }
      if (t + 2 < ntiles) {   // pull tile t+2 toward GL2
        const int k2 = (t + 2) << 5;
        if (A_BF16)
          __builtin_prefetch((const __bf16*)Aptr + (size_t)(m0 + am[0]) * lda + k2 + ak[0], 0, 1);
        else
          __builtin_prefetch((const float*)Aptr + (size_t)(m0 + am[0]) * lda + k2 + ak[0], 0, 1);
        __builtin_prefetch(Bw + (size_t)(k2 + bk[0]) * ldb + n0 + bn[0], 0, 1);
      }
    }
    // ---- compute tile t: load ALL fragments, then WMMA burst ----
    if (t >= 0) {
      const int cur = t & 1;
      v16bf bfrag[2], afrag[4];
#pragma unroll
      for (int nf = 0; nf < 2; ++nf) {
        const __bf16* p = &Bs[cur][wn * 32 + nf * 16 + l16][half * 8];
        bfrag[nf] = ld16x2(p, p + 16);
      }
#pragma unroll
      for (int mf = 0; mf < 4; ++mf) {
        const __bf16* pa = &As[cur][wm * 64 + mf * 16 + l16][half * 8];
        afrag[mf] = ld16x2(pa, pa + 16);
      }
#pragma unroll
      for (int mf = 0; mf < 4; ++mf)
#pragma unroll
        for (int nf = 0; nf < 2; ++nf)
          acc[mf][nf] = wmma_bf16(afrag[mf], bfrag[nf], acc[mf][nf]);
    }
    // ---- commit staged tile t+1 into the other LDS buffer ----
    if (more) {
      const int nxt = (t + 1) & 1;
#pragma unroll
      for (int j = 0; j < 4; ++j) {
        v4bf o;
        if (A_BF16) o = rab[j];
        else { o[0] = (__bf16)raf[j].x; o[1] = (__bf16)raf[j].y;
               o[2] = (__bf16)raf[j].z; o[3] = (__bf16)raf[j].w; }
        *(v4bf*)(&As[nxt][am[j]][ak[j]]) = o;
        Bs[nxt][bn[j] + 0][bk[j]] = (__bf16)rbf[j].x;
        Bs[nxt][bn[j] + 1][bk[j]] = (__bf16)rbf[j].y;
        Bs[nxt][bn[j] + 2][bk[j]] = (__bf16)rbf[j].z;
        Bs[nxt][bn[j] + 3][bk[j]] = (__bf16)rbf[j].w;
      }
    }
    __syncthreads();
  }

  // ---- store C: VGPR r holds rows (r, r+8), N = l16 ----
#pragma unroll
  for (int mf = 0; mf < 4; ++mf)
#pragma unroll
    for (int nf = 0; nf < 2; ++nf) {
      int col = n0 + wn * 32 + nf * 16 + l16;
#pragma unroll
      for (int r = 0; r < 8; ++r) {
        int row = m0 + wm * 64 + mf * 16 + r + half * 8;
        float v = acc[mf][nf][r];
        if (OUT_BF16) ((__bf16*)Cptr)[(size_t)row * ldc + col] = (__bf16)v;
        else          ((float*)Cptr)[(size_t)row * ldc + col]  = v;
      }
    }
}

// =====================================================================
// Flash attention (causal). One block = 128 q rows of one (b,h).
// 8 waves x 16 rows; KV tile = 64 (softmax cost amortized over 2x the
// WMMA work); K/V double-buffered; exp2-domain online softmax.
// =====================================================================
__global__ __launch_bounds__(256) void attn_fa_k(
    const __bf16* __restrict__ qkv, __bf16* __restrict__ y) {
  __shared__ __align__(16) __bf16 Ks[2][64][136];     // [buf][kv][d]
  __shared__ __align__(16) __bf16 Vt[2][128][72];     // [buf][d][kv]
  __shared__ __align__(16) __bf16 Ps[8][16][72];      // per-wave P tile (16x64)

  const int b  = blockIdx.z;
  const int h  = blockIdx.y;
  const int q0 = blockIdx.x * 128;
  const int tid = threadIdx.x, lane = tid & 31, wave = tid >> 5;
  const int l16 = lane & 15, half = lane >> 4;

  const size_t rs = 3 * CC;  // qkv row stride (elements)
  const __bf16* qb = qkv + (size_t)b * TT * rs + h * DH;
  const __bf16* kb = qb + CC;
  const __bf16* vb = qb + 2 * CC;

  // ---- load Q fragments (16 rows x 128 d = 4 A-frags) from global ----
  const int qrow = q0 + wave * 16 + l16;
  v16bf qf[4];
#pragma unroll
  for (int kf = 0; kf < 4; ++kf) {
    const __bf16* p = qb + (size_t)qrow * rs + kf * 32 + half * 8;
    qf[kf] = ld16x2(p, p + 16);
  }

  v8f o[8] = {};
  float mrow[8], lrow[8];
#pragma unroll
  for (int r = 0; r < 8; ++r) { mrow[r] = -1e30f; lrow[r] = 0.0f; }
  const float scale2 = 0.08838834764831845f * 1.44269504088896340f;  // 1/sqrt(d)*log2e
  const int row_lo = q0 + wave * 16;

  const int srow = tid >> 2;            // staging row 0..63
  const int sd0  = (tid & 3) * 32;      // staging d offset 0/32/64/96
  const int ntiles = (q0 + 128) >> 6;   // 64-wide KV tiles

  v8bf kreg[4], vreg[4];

  for (int t = -1; t < ntiles; ++t) {
    const bool more = (t + 1) < ntiles;
    // ---- issue global loads for K/V tile t+1 ----
    if (more) {
      const __bf16* kp = kb + (size_t)(((t + 1) << 6) + srow) * rs + sd0;
      const __bf16* vp = vb + (size_t)(((t + 1) << 6) + srow) * rs + sd0;
#pragma unroll
      for (int i = 0; i < 4; ++i) {
        kreg[i] = *(const v8bf*)(kp + 8 * i);
        vreg[i] = *(const v8bf*)(vp + 8 * i);
      }
      if (t + 2 < ntiles) {
        __builtin_prefetch(kb + (size_t)(((t + 2) << 6) + srow) * rs + sd0, 0, 1);
        __builtin_prefetch(vb + (size_t)(((t + 2) << 6) + srow) * rs + sd0, 0, 1);
      }
    }
    // ---- compute tile t ----
    if (t >= 0) {
      const int cur = t & 1;
      const int kv0 = t << 6;
      // S = Q K^T : 4 column-fragments x 4 K-steps = 16 WMMAs
      v8f s[4] = {};
#pragma unroll
      for (int c = 0; c < 4; ++c) {
        v16bf kfrag[4];
#pragma unroll
        for (int kf = 0; kf < 4; ++kf) {
          const __bf16* p = &Ks[cur][c * 16 + l16][kf * 32 + half * 8];
          kfrag[kf] = ld16x2(p, p + 16);
        }
#pragma unroll
        for (int kf = 0; kf < 4; ++kf)
          s[c] = wmma_bf16(qf[kf], kfrag[kf], s[c]);
      }
#pragma unroll
      for (int c = 0; c < 4; ++c)
#pragma unroll
        for (int r = 0; r < 8; ++r) s[c][r] *= scale2;

      // causal mask (wave-uniform outer branch)
      if (kv0 + 63 > row_lo) {
#pragma unroll
        for (int r = 0; r < 8; ++r) {
          int row = row_lo + r + half * 8;
#pragma unroll
          for (int c = 0; c < 4; ++c)
            if (kv0 + c * 16 + l16 > row) s[c][r] = -1e30f;
        }
      }

      // online softmax in exp2 domain (16-lane group = one row)
#pragma unroll
      for (int r = 0; r < 8; ++r) {
        float mx = fmaxf(fmaxf(s[0][r], s[1][r]), fmaxf(s[2][r], s[3][r]));
#pragma unroll
        for (int off = 8; off >= 1; off >>= 1) mx = fmaxf(mx, __shfl_xor(mx, off, 32));
        float mnew  = fmaxf(mrow[r], mx);
        float alpha = __builtin_amdgcn_exp2f(mrow[r] - mnew);
        float p0 = __builtin_amdgcn_exp2f(s[0][r] - mnew);
        float p1 = __builtin_amdgcn_exp2f(s[1][r] - mnew);
        float p2 = __builtin_amdgcn_exp2f(s[2][r] - mnew);
        float p3 = __builtin_amdgcn_exp2f(s[3][r] - mnew);
        float sum = (p0 + p1) + (p2 + p3);
#pragma unroll
        for (int off = 8; off >= 1; off >>= 1) sum += __shfl_xor(sum, off, 32);
        lrow[r] = lrow[r] * alpha + sum;
        mrow[r] = mnew;
#pragma unroll
        for (int df = 0; df < 8; ++df) o[df][r] *= alpha;
        int prow = r + half * 8;
        Ps[wave][prow][l16]      = (__bf16)p0;
        Ps[wave][prow][16 + l16] = (__bf16)p1;
        Ps[wave][prow][32 + l16] = (__bf16)p2;
        Ps[wave][prow][48 + l16] = (__bf16)p3;
      }

      // reload P (16x64) as 2 A-fragments (per-wave LDS, in-order DS)
      const __bf16* pp = &Ps[wave][l16][half * 8];
      v16bf pf0 = ld16x2(pp, pp + 16);
      v16bf pf1 = ld16x2(pp + 32, pp + 48);

      // O += P @ V : 8 d-chunks x 2 K-steps = 16 WMMAs
#pragma unroll
      for (int df = 0; df < 8; ++df) {
        const __bf16* va = &Vt[cur][df * 16 + l16][half * 8];
        v16bf vf0 = ld16x2(va, va + 16);
        v16bf vf1 = ld16x2(va + 32, va + 48);
        o[df] = wmma_bf16(pf0, vf0, o[df]);
        o[df] = wmma_bf16(pf1, vf1, o[df]);
      }
    }
    // ---- commit staged K/V tile t+1 ----
    if (more) {
      const int nxt = (t + 1) & 1;
#pragma unroll
      for (int i = 0; i < 4; ++i)
        *(v8bf*)(&Ks[nxt][srow][sd0 + 8 * i]) = kreg[i];
      __bf16 vtmp[32];
#pragma unroll
      for (int i = 0; i < 4; ++i) *(v8bf*)(vtmp + 8 * i) = vreg[i];
#pragma unroll
      for (int i = 0; i < 32; ++i) Vt[nxt][sd0 + i][srow] = vtmp[i];
    }
    __syncthreads();
  }

  // ---- epilogue: normalize and store y (bf16) ----
  __bf16* yb = y + (size_t)b * TT * CC + h * DH;
#pragma unroll
  for (int df = 0; df < 8; ++df) {
    int col = df * 16 + l16;
#pragma unroll
    for (int r = 0; r < 8; ++r) {
      int row = q0 + wave * 16 + r + half * 8;
      yb[(size_t)row * CC + col] = (__bf16)(o[df][r] / lrow[r]);
    }
  }
}

// =====================================================================
extern "C" void kernel_launch(void* const* d_in, const int* in_sizes, int n_in,
                              void* d_out, int out_size, void* d_ws, size_t ws_size,
                              hipStream_t stream) {
  const float* x  = (const float*)d_in[0];   // (B,T,C) f32
  const float* Wa = (const float*)d_in[1];   // (C,3C)  f32
  const float* Wp = (const float*)d_in[2];   // (C,C)   f32
  float* out = (float*)d_out;                // (B,T,C) f32

  const int M = BB * TT;           // 4096
  __bf16* qkv = (__bf16*)d_ws;                                   // M x 3C bf16
  __bf16* y   = (__bf16*)((char*)d_ws + (size_t)M * 3 * CC * 2); // M x C  bf16

  dim3 blk(256);
  // 1) qkv = x @ W_attn  (f32 in, bf16 out)
  gemm_wmma_k<false, true><<<dim3((3 * CC) / 128, M / 128), blk, 0, stream>>>(
      x, Wa, qkv, M, 3 * CC, CC, CC, 3 * CC, 3 * CC);
  // 2) flash attention -> y (bf16)
  attn_fa_k<<<dim3(TT / 128, NH, BB), blk, 0, stream>>>(qkv, y);
  // 3) out = y @ W_proj (bf16 in, f32 out)
  gemm_wmma_k<true, false><<<dim3(CC / 128, M / 128), blk, 0, stream>>>(
      y, Wp, out, M, CC, CC, CC, CC, CC);
}
